// VisionBlock_17901423690403
// MI455X (gfx1250) — compile-verified, matrix operand order checked by reading, not run on hardware
//
#include <hip/hip_runtime.h>
#include <hip/hip_bf16.h>

#define S_     16384
#define HID_   1152
#define NH_    16
#define HD_    72
#define INTER_ 4304
#define NP_    4352      // INTER padded to a multiple of 128 (and 32)
#define N3_    3456
#define NWIN_  256

typedef __attribute__((ext_vector_type(16))) __bf16 v16bf;
typedef __attribute__((ext_vector_type(8)))  __bf16 v8bf;
typedef __attribute__((ext_vector_type(8)))  float  v8f;

__device__ __forceinline__ unsigned short f2bf(float f) {
  union { float f; unsigned u; } v; v.f = f;
  unsigned r = v.u + 0x7FFFu + ((v.u >> 16) & 1u);   // round-to-nearest-even
  return (unsigned short)(r >> 16);
}
__device__ __forceinline__ float bf2f(unsigned short u) {
  union { unsigned u; float f; } v; v.u = ((unsigned)u) << 16;
  return v.f;
}

// ---------------------------------------------------------------------------
// Weight convert + transpose + pad: W[K,N] f32 -> WT[Np,Kp] bf16, zero-filled
// outside [N,K]. WT rows are K-contiguous so WMMA A/B fragments are two
// contiguous 16B LDS loads. Writes coalesced (k fastest).
// ---------------------------------------------------------------------------
__global__ __launch_bounds__(256) void convT_kernel(const float* __restrict__ W,
                                                    unsigned short* __restrict__ WT,
                                                    int K, int N, int Kp, int Np) {
  long long idx = (long long)blockIdx.x * 256 + threadIdx.x;
  long long tot = (long long)Kp * Np;
  if (idx >= tot) return;
  int k = (int)(idx % Kp);
  int n = (int)(idx / Kp);
  unsigned short v = 0;
  if (k < K && n < N) v = f2bf(W[(long long)k * N + n]);
  WT[idx] = v;
}

__global__ __launch_bounds__(256) void padbias_kernel(const float* __restrict__ b,
                                                      float* __restrict__ bp,
                                                      int N, int Np) {
  int i = blockIdx.x * 256 + threadIdx.x;
  if (i < Np) bp[i] = (i < N) ? b[i] : 0.f;
}

// ---------------------------------------------------------------------------
// LayerNorm: one block (256 thr = 8 waves) per row; f32 in, bf16 out.
// ---------------------------------------------------------------------------
__global__ __launch_bounds__(256) void ln_kernel(const float* __restrict__ x,
                                                 const float* __restrict__ w,
                                                 const float* __restrict__ b,
                                                 unsigned short* __restrict__ out,
                                                 int H) {
  int row = blockIdx.x;
  const float* xr = x + (size_t)row * H;
  __shared__ float ps[8], ps2[8];
  float s = 0.f, s2 = 0.f;
  for (int i = threadIdx.x; i < H; i += 256) { float v = xr[i]; s += v; s2 += v * v; }
  for (int o = 16; o > 0; o >>= 1) { s += __shfl_down(s, o, 32); s2 += __shfl_down(s2, o, 32); }
  int wave = threadIdx.x >> 5, lane = threadIdx.x & 31;
  if (lane == 0) { ps[wave] = s; ps2[wave] = s2; }
  __syncthreads();
  float ts = 0.f, ts2 = 0.f;
#pragma unroll
  for (int i = 0; i < 8; i++) { ts += ps[i]; ts2 += ps2[i]; }
  float mean = ts / H;
  float var  = ts2 / H - mean * mean;
  float rstd = rsqrtf(var + 1e-6f);
  unsigned short* orow = out + (size_t)row * H;
  for (int i = threadIdx.x; i < H; i += 256)
    orow[i] = f2bf((xr[i] - mean) * rstd * w[i] + b[i]);
}

// ---------------------------------------------------------------------------
// bf16 WMMA GEMM: C[M,N] = A[M,K] @ BT[N,K]^T (+bias, optional GELU/residual).
// All dims pre-padded: M%128==0, N%128==0, K%32==0 -> no bounds checks, EXEC
// stays all-ones. 256 thr = 8 wave32, block tile 128x128, wave tile 32x64.
// Double-buffered LDS software pipeline: global b128 loads for tile kt+1 fly
// while tile kt is consumed by ds_load_b128 + v_wmma_f32_16x16x32_bf16.
// ---------------------------------------------------------------------------
template <bool GELU, bool RESID, bool OUTBF>
__global__ __launch_bounds__(256) void gemm_bf16_kernel(
    const unsigned short* __restrict__ A,    // M x K bf16 (row major)
    const unsigned short* __restrict__ BT,   // N x K bf16 (row major = B^T)
    const float* __restrict__ bias,          // N
    const float* __restrict__ resid,         // M x N f32 (if RESID)
    float* __restrict__ outF,                // if !OUTBF
    unsigned short* __restrict__ outB,       // if OUTBF
    int N, int K) {
  // LDS tiles: rows padded to 40 elems (80B) to spread banks. 2x double buffer.
  __shared__ __bf16 lA[2][128 * 40];
  __shared__ __bf16 lB[2][128 * 40];

  const int tid   = threadIdx.x;
  const int wave  = tid >> 5;
  const int lane  = tid & 31;
  const int laneR = lane & 15;       // row (A) / col (B) within 16x16 tile
  const int laneH = lane >> 4;       // kbase = laneH*8; C row offset +8*laneH
  const int waveM = wave >> 1;       // 0..3
  const int waveN = wave & 1;        // 0..1
  const int rowBase = blockIdx.x * 128;
  const int colBase = blockIdx.y * 128;

  // staging: thread owns chunk (r0,c0) and (r0+64,c0); 8 bf16 = 16B per chunk
  const int r0 = tid >> 2;           // 0..63
  const int c0 = (tid & 3) * 8;      // 0,8,16,24
  const unsigned short* Ar0 = A  + (size_t)(rowBase + r0)      * K + c0;
  const unsigned short* Ar1 = A  + (size_t)(rowBase + r0 + 64) * K + c0;
  const unsigned short* Br0 = BT + (size_t)(colBase + r0)      * K + c0;
  const unsigned short* Br1 = BT + (size_t)(colBase + r0 + 64) * K + c0;
  const int lo0 = r0 * 40 + c0;
  const int lo1 = (r0 + 64) * 40 + c0;

  v8f acc[2][4];
#pragma unroll
  for (int m = 0; m < 2; m++)
#pragma unroll
    for (int n = 0; n < 4; n++) acc[m][n] = (v8f)0.f;

  const int ksteps = K >> 5;

  // prologue: stage tile 0 into buffer 0
  uint4 ra0 = *(const uint4*)(Ar0);
  uint4 ra1 = *(const uint4*)(Ar1);
  uint4 rb0 = *(const uint4*)(Br0);
  uint4 rb1 = *(const uint4*)(Br1);
  *(uint4*)(&lA[0][lo0]) = ra0;
  *(uint4*)(&lA[0][lo1]) = ra1;
  *(uint4*)(&lB[0][lo0]) = rb0;
  *(uint4*)(&lB[0][lo1]) = rb1;

  int p = 0;
  for (int kt = 0; kt < ksteps; kt++) {
    __syncthreads();                     // buffer p fully written
    const int kn = (kt + 1) << 5;
    if (kt + 1 < ksteps) {               // launch next tile's global loads
      ra0 = *(const uint4*)(Ar0 + kn);
      ra1 = *(const uint4*)(Ar1 + kn);
      rb0 = *(const uint4*)(Br0 + kn);
      rb1 = *(const uint4*)(Br1 + kn);
      if (kt + 2 < ksteps) {             // pull kt+2 toward L2/L0
        __builtin_prefetch(Ar0 + kn + 32, 0, 1);
        __builtin_prefetch(Br0 + kn + 32, 0, 1);
      }
    }

    // fragments from LDS (ISA 16-bit A layout: lane=row laneR, kbase=laneH*8,
    // elems 0..7 @ kbase, elems 8..15 @ kbase+16)
    const int kb = laneH * 8;
    const __bf16* bufA = lA[p];
    const __bf16* bufB = lB[p];
    v16bf fa[2], fb[4];
#pragma unroll
    for (int m = 0; m < 2; m++) {
      int r = (waveM * 32 + m * 16 + laneR) * 40 + kb;
      v8bf lo = *(const v8bf*)(&bufA[r]);
      v8bf hi = *(const v8bf*)(&bufA[r + 16]);
      v16bf f;
#pragma unroll
      for (int e = 0; e < 8; e++) { f[e] = lo[e]; f[e + 8] = hi[e]; }
      fa[m] = f;
    }
#pragma unroll
    for (int n = 0; n < 4; n++) {
      int r = (waveN * 64 + n * 16 + laneR) * 40 + kb;
      v8bf lo = *(const v8bf*)(&bufB[r]);
      v8bf hi = *(const v8bf*)(&bufB[r + 16]);
      v16bf f;
#pragma unroll
      for (int e = 0; e < 8; e++) { f[e] = lo[e]; f[e + 8] = hi[e]; }
      fb[n] = f;
    }
#pragma unroll
    for (int m = 0; m < 2; m++)
#pragma unroll
      for (int n = 0; n < 4; n++)
        acc[m][n] = __builtin_amdgcn_wmma_f32_16x16x32_bf16(
            false, fa[m], false, fb[n], (short)0, acc[m][n], false, false);

    if (kt + 1 < ksteps) {               // commit next tile into other buffer
      *(uint4*)(&lA[p ^ 1][lo0]) = ra0;
      *(uint4*)(&lA[p ^ 1][lo1]) = ra1;
      *(uint4*)(&lB[p ^ 1][lo0]) = rb0;
      *(uint4*)(&lB[p ^ 1][lo1]) = rb1;
    }
    p ^= 1;
  }

  // epilogue: C layout = row m*16 + i + 8*laneH, col laneR (v8f element i)
#pragma unroll
  for (int m = 0; m < 2; m++) {
#pragma unroll
    for (int n = 0; n < 4; n++) {
      int col = colBase + waveN * 64 + n * 16 + laneR;
      float bv = bias[col];
#pragma unroll
      for (int i = 0; i < 8; i++) {
        int row = rowBase + waveM * 32 + m * 16 + laneH * 8 + i;
        float v = acc[m][n][i] + bv;
        if (GELU) {
          float t = v;
          v = 0.5f * t * (1.f + tanhf(0.7978845608028654f * (t + 0.044715f * t * t * t)));
        }
        if (RESID) v += resid[(size_t)row * N + col];
        if (OUTBF) outB[(size_t)row * N + col] = f2bf(v);
        else       outF[(size_t)row * N + col] = v;
      }
    }
  }
}

// ---------------------------------------------------------------------------
// Windowed attention: one block per (window, head), 64 threads = 1 query row
// each. RoPE(HD=72, halves of 36) on q/k, f32 softmax, bf16 out for proj GEMM.
// ---------------------------------------------------------------------------
__global__ __launch_bounds__(64) void attn_kernel(
    const unsigned short* __restrict__ qkv,  // [S, 3*HID] bf16 (q|k|v)
    const float* __restrict__ cosp,          // [S, 72]
    const float* __restrict__ sinp,          // [S, 72]
    unsigned short* __restrict__ out) {      // [S, HID] bf16
  const int win  = blockIdx.x;
  const int head = blockIdx.y;
  const int t    = threadIdx.x;              // query/key row 0..63
  const int pos  = win * 64 + t;

  __shared__ float Ks[64][HD_ + 1];
  __shared__ float Vs[64][HD_ + 1];

  const unsigned short* base = qkv + (size_t)pos * (3 * HID_) + head * HD_;
  const float* cr = cosp + (size_t)pos * HD_;
  const float* sr = sinp + (size_t)pos * HD_;

  float q[HD_];
  float kr[HD_];
#pragma unroll
  for (int d = 0; d < HD_; d++) {
    q[d]  = bf2f(base[d]);
    kr[d] = bf2f(base[HID_ + d]);
    Vs[t][d] = bf2f(base[2 * HID_ + d]);
  }
  // RoPE: y[d] = x[d]*cos[d] + rot(x)[d]*sin[d], rot = (-x[36:], x[:36])
#pragma unroll
  for (int d = 0; d < HD_ / 2; d++) {
    float c0 = cr[d], s0 = sr[d], c1 = cr[d + 36], s1 = sr[d + 36];
    float q0 = q[d], q1 = q[d + 36];
    q[d]      = q0 * c0 - q1 * s0;
    q[d + 36] = q1 * c1 + q0 * s1;
    float k0 = kr[d], k1 = kr[d + 36];
    Ks[t][d]      = k0 * c0 - k1 * s0;
    Ks[t][d + 36] = k1 * c1 + k0 * s1;
  }
  __syncthreads();

  float sc[64];
  float mx = -1e30f;
#pragma unroll 4
  for (int j = 0; j < 64; j++) {
    float a = 0.f;
#pragma unroll
    for (int d = 0; d < HD_; d++) a += q[d] * Ks[j][d];
    a *= 0.117851130197758f;                 // 1/sqrt(72)
    sc[j] = a;
    mx = fmaxf(mx, a);
  }
  float sum = 0.f;
#pragma unroll 4
  for (int j = 0; j < 64; j++) { sc[j] = __expf(sc[j] - mx); sum += sc[j]; }
  float inv = 1.f / sum;

  unsigned short* orow = out + (size_t)pos * HID_ + head * HD_;
#pragma unroll 2
  for (int d = 0; d < HD_; d++) {
    float a = 0.f;
#pragma unroll 8
    for (int j = 0; j < 64; j++) a += sc[j] * Vs[j][d];
    orow[d] = f2bf(a * inv);
  }
}

// ---------------------------------------------------------------------------
extern "C" void kernel_launch(void* const* d_in, const int* in_sizes, int n_in,
                              void* d_out, int out_size, void* d_ws, size_t ws_size,
                              hipStream_t stream) {
  (void)in_sizes; (void)n_in; (void)out_size; (void)ws_size;
  const float* x      = (const float*)d_in[0];
  const float* cosp   = (const float*)d_in[1];
  const float* sinp   = (const float*)d_in[2];
  /* d_in[3] = cu_seqlens (uniform 64-token windows; not needed) */
  const float* qkv_w  = (const float*)d_in[4];
  const float* qkv_b  = (const float*)d_in[5];
  const float* proj_w = (const float*)d_in[6];
  const float* proj_b = (const float*)d_in[7];
  const float* ln1_w  = (const float*)d_in[8];
  const float* ln1_b  = (const float*)d_in[9];
  const float* ln2_w  = (const float*)d_in[10];
  const float* ln2_b  = (const float*)d_in[11];
  const float* fc1_w  = (const float*)d_in[12];
  const float* fc1_b  = (const float*)d_in[13];
  const float* fc2_w  = (const float*)d_in[14];
  const float* fc2_b  = (const float*)d_in[15];
  float* out = (float*)d_out;

  char* ws = (char*)d_ws;
  size_t off = 0;
  auto take = [&](size_t bytes) -> char* {
    char* p = ws + off;
    off += (bytes + 255) & ~(size_t)255;
    return p;
  };
  unsigned short* wqkvT  = (unsigned short*)take((size_t)N3_ * HID_ * 2);
  unsigned short* wprojT = (unsigned short*)take((size_t)HID_ * HID_ * 2);
  unsigned short* wfc1T  = (unsigned short*)take((size_t)NP_ * HID_ * 2);   // padded N
  unsigned short* wfc2T  = (unsigned short*)take((size_t)HID_ * NP_ * 2);   // padded K
  float*          b1pad  = (float*)take((size_t)NP_ * 4);
  unsigned short* Abuf   = (unsigned short*)take((size_t)S_ * HID_ * 2);    // LN1, later LN2
  unsigned short* Big    = (unsigned short*)take((size_t)S_ * NP_ * 2);     // QKV, later FC1
  unsigned short* Attn   = (unsigned short*)take((size_t)S_ * HID_ * 2);
  float*          X2     = (float*)take((size_t)S_ * HID_ * 4);

  auto cgrid = [](long long tot) { return dim3((unsigned)((tot + 255) / 256)); };

  // --- weight transpose+convert to bf16 [Np,Kp], zero padded
  convT_kernel<<<cgrid((long long)HID_ * N3_), 256, 0, stream>>>(qkv_w,  wqkvT,  HID_,   N3_,  HID_, N3_);
  convT_kernel<<<cgrid((long long)HID_ * HID_),256, 0, stream>>>(proj_w, wprojT, HID_,   HID_, HID_, HID_);
  convT_kernel<<<cgrid((long long)HID_ * NP_), 256, 0, stream>>>(fc1_w,  wfc1T,  HID_, INTER_, HID_, NP_);
  convT_kernel<<<cgrid((long long)NP_ * HID_), 256, 0, stream>>>(fc2_w,  wfc2T,  INTER_, HID_, NP_,  HID_);
  padbias_kernel<<<cgrid(NP_), 256, 0, stream>>>(fc1_b, b1pad, INTER_, NP_);

  // --- LN1 -> bf16 A
  ln_kernel<<<dim3(S_), 256, 0, stream>>>(x, ln1_w, ln1_b, Abuf, HID_);

  // --- QKV GEMM: [S,1152] x [1152,3456] -> bf16
  gemm_bf16_kernel<false, false, true><<<dim3(S_ / 128, N3_ / 128), 256, 0, stream>>>(
      Abuf, wqkvT, qkv_b, nullptr, nullptr, Big, N3_, HID_);

  // --- windowed attention with RoPE -> bf16
  attn_kernel<<<dim3(NWIN_, NH_), 64, 0, stream>>>(Big, cosp, sinp, Attn);

  // --- proj GEMM + residual(x) -> f32 X2
  gemm_bf16_kernel<false, true, false><<<dim3(S_ / 128, HID_ / 128), 256, 0, stream>>>(
      Attn, wprojT, proj_b, x, X2, nullptr, HID_, HID_);

  // --- LN2 -> bf16 A
  ln_kernel<<<dim3(S_), 256, 0, stream>>>(X2, ln2_w, ln2_b, Abuf, HID_);

  // --- FC1 GEMM + GELU -> bf16 [S, NP_] (padded cols come out exactly 0)
  gemm_bf16_kernel<true, false, true><<<dim3(S_ / 128, NP_ / 128), 256, 0, stream>>>(
      Abuf, wfc1T, b1pad, nullptr, nullptr, Big, NP_, HID_);

  // --- FC2 GEMM (K = NP_, zero-padded rows of W2^T) + residual(X2) -> d_out
  gemm_bf16_kernel<false, true, false><<<dim3(S_ / 128, HID_ / 128), 256, 0, stream>>>(
      Big, wfc2T, fc2_b, X2, out, nullptr, HID_, NP_);
}
